// MultiHeadAttention_10514079941592
// MI455X (gfx1250) — matrix operand order repaired and checked by reading, not compiled, not run on hardware
//
#include <hip/hip_runtime.h>
#include <hip/hip_bf16.h>
#include <math.h>

// ---------------------------------------------------------------------------
// MHA forward for MI455X (gfx1250, wave32, WMMA 16x16x32 f16, f32 accum).
// S=4096, D=768, H=12, DK=64.
// ---------------------------------------------------------------------------

#define S_LEN 4096
#define D_MODEL 768
#define NH 12
#define DKH 64

typedef _Float16 v16h __attribute__((ext_vector_type(16)));
typedef float    v8f  __attribute__((ext_vector_type(8)));
typedef int      v4i  __attribute__((ext_vector_type(4)));

union AFrag { v16h v; _Float16 h[16]; uint4 q[2]; };
union CFrag { v8f  v; float f[8]; };

__device__ __forceinline__ v8f wmma_f16(v16h a, v16h b, v8f c) {
  // (neg_a, A, neg_b, B, c_mod, C, reuse_a, reuse_b)
  return __builtin_amdgcn_wmma_f32_16x16x32_f16(false, a, false, b, (short)0, c,
                                                false, false);
}

// convert 8 consecutive f32 -> 8 f16
__device__ __forceinline__ void cvt8(_Float16* d, const float* s) {
  float4 a = *reinterpret_cast<const float4*>(s);
  float4 b = *reinterpret_cast<const float4*>(s + 4);
  d[0] = (_Float16)a.x; d[1] = (_Float16)a.y; d[2] = (_Float16)a.z; d[3] = (_Float16)a.w;
  d[4] = (_Float16)b.x; d[5] = (_Float16)b.y; d[6] = (_Float16)b.z; d[7] = (_Float16)b.w;
}

// ---- gfx1250 async global->LDS copy (ASYNCcnt) with sync b128 fallback -----
#if defined(__has_builtin)
#if __has_builtin(__builtin_amdgcn_global_load_async_to_lds_b128) && \
    __has_builtin(__builtin_amdgcn_s_wait_asynccnt)
#define USE_ASYNC_LDS 1
#endif
#endif

typedef __attribute__((address_space(1))) v4i* gv4i_p;  // global int4*
typedef __attribute__((address_space(3))) v4i* lv4i_p;  // LDS int4*

__device__ __forceinline__ void g2l_b128(_Float16* ldst, const _Float16* gsrc) {
#ifdef USE_ASYNC_LDS
  // generic LDS pointer: low 32 bits are the LDS byte offset (aperture rule);
  // generic global pointer: identical to as(1) address.
  gv4i_p g = (gv4i_p)(uintptr_t)gsrc;
  lv4i_p l = (lv4i_p)(unsigned)(uintptr_t)ldst;
  __builtin_amdgcn_global_load_async_to_lds_b128(g, l, 0, 0);
#else
  *reinterpret_cast<uint4*>(ldst) = *reinterpret_cast<const uint4*>(gsrc);
#endif
}

// ---------------------------------------------------------------------------
// Projection: Y_f16 = (X_f32 @ W_f32 + b) * alpha, written per-head.
// mode 0: Y[h][s][d]   (Q with alpha=1/8, V with alpha=1)
// mode 1: Y[h][d][s]   (K transposed, so QK^T B-frags are key-contiguous)
// ---------------------------------------------------------------------------
__global__ __launch_bounds__(128) void proj_f16_kernel(
    const float* __restrict__ X, const float* __restrict__ W,
    const float* __restrict__ bias, _Float16* __restrict__ Y,
    float alpha, int mode) {
  __shared__ alignas(16) _Float16 Bs[32 * 72];   // 32 x 64 (+8 pad)

  const int m0 = blockIdx.x * 64;
  const int n0 = blockIdx.y * 64;
  const int t  = threadIdx.x;
  const int lane = t & 31, w = t >> 5;
  const int ml = lane & 15, hl = lane >> 4;
  const int mrow = m0 + w * 16 + ml;

  CFrag c[4];
#pragma unroll
  for (int i = 0; i < 4; ++i)
#pragma unroll
    for (int j = 0; j < 8; ++j) c[i].f[j] = 0.0f;

  for (int k0 = 0; k0 < D_MODEL; k0 += 32) {
    __syncthreads();
#pragma unroll
    for (int i = 0; i < 2; ++i) {
      int u = t + 128 * i;
      int r = u >> 3, c8 = (u & 7) * 8;
      AFrag tmp;
      cvt8(tmp.h, W + (size_t)(k0 + r) * D_MODEL + n0 + c8);
      *reinterpret_cast<uint4*>(&Bs[r * 72 + c8]) = tmp.q[0];
    }
    __syncthreads();

    // A fragment: row M=ml, halves e: K = 8*hl + (e&7) + 16*(e>>3)
    AFrag a;
    const float* Xr = X + (size_t)mrow * D_MODEL + k0 + 8 * hl;
    cvt8(a.h, Xr);
    cvt8(a.h + 8, Xr + 16);

#pragma unroll
    for (int nt = 0; nt < 4; ++nt) {
      AFrag b;  // lane L holds B col N=L%16, K = 8*hl + (e&7) + 16*(e>>3)
      const _Float16* bp = &Bs[nt * 16 + ml];
      for (int e = 0; e < 16; ++e) b.h[e] = bp[(8 * hl + (e & 7) + 16 * (e >> 3)) * 72];
      c[nt].v = wmma_f16(a.v, b.v, c[nt].v);
    }
  }

#pragma unroll
  for (int nt = 0; nt < 4; ++nt) {
    int n = n0 + nt * 16 + ml;          // C layout: lane's column = ml
    float bv = bias[n];
    int head = n >> 6, d = n & 63;
#pragma unroll
    for (int j = 0; j < 8; ++j) {       // row M = j + 8*hl
      int srow = m0 + w * 16 + j + 8 * hl;
      float val = (c[nt].f[j] + bv) * alpha;
      size_t off = (mode == 0)
          ? ((size_t)head * S_LEN * DKH + (size_t)srow * DKH + d)
          : ((size_t)head * DKH * S_LEN + (size_t)d * S_LEN + srow);
      Y[off] = (_Float16)val;
    }
  }
}

// ---------------------------------------------------------------------------
// Flash attention. Block = 128 thr = 4 waves on one head. Each wave owns TWO
// 16-query tiles (32 queries) so every K/V B-fragment feeds two WMMAs.
// K/V tiles (32 keys) are double-buffered in LDS; next tile is issued
// (async-to-LDS when available) before computing the current one.
// Q is pre-scaled by 1/sqrt(DK). Scores never touch global memory.
// ---------------------------------------------------------------------------
__device__ __forceinline__ void stage_kv(_Float16* __restrict__ KsS,
                                         _Float16* __restrict__ VsS,
                                         const _Float16* __restrict__ Ktb,
                                         const _Float16* __restrict__ Vb,
                                         int kb, int t) {
#pragma unroll
  for (int i = 0; i < 2; ++i) {
    int u = t + 128 * i;
    {
      int r = u >> 2, c8 = (u & 3) * 8;   // K^T tile: 64 d x 32 keys
      g2l_b128(&KsS[r * 40 + c8], Ktb + (size_t)r * S_LEN + kb + c8);
    }
    {
      int r = u >> 3, c8 = (u & 7) * 8;   // V tile: 32 keys x 64 d
      g2l_b128(&VsS[r * 72 + c8], Vb + (size_t)(kb + r) * DKH + c8);
    }
  }
}

__global__ __launch_bounds__(128) void attn_kernel(
    const _Float16* __restrict__ Qh,   // [H][S][64], scaled
    const _Float16* __restrict__ Kt,   // [H][64][S]
    const _Float16* __restrict__ Vh,   // [H][S][64]
    _Float16* __restrict__ Ctx) {      // [S][768]
  __shared__ alignas(16) _Float16 Ks[2][64 * 40];     // double-buffered
  __shared__ alignas(16) _Float16 Vs[2][32 * 72];
  __shared__ alignas(16) _Float16 Ps[4][2][16 * 40];  // per-wave, per-qtile P

  const int head = blockIdx.y;
  const int m0   = blockIdx.x * 128;
  const int t = threadIdx.x, lane = t & 31, w = t >> 5;
  const int ml = lane & 15, hl = lane >> 4;

  // Resident Q A-fragments: 2 query tiles x 2 d-chunks
  AFrag qa[2][2];
#pragma unroll
  for (int qt = 0; qt < 2; ++qt) {
    int qrow = m0 + (w * 2 + qt) * 16 + ml;
    const _Float16* Qb = Qh + (size_t)head * S_LEN * DKH + (size_t)qrow * DKH;
#pragma unroll
    for (int dc = 0; dc < 2; ++dc) {
      qa[qt][dc].q[0] = *reinterpret_cast<const uint4*>(Qb + dc * 32 + 8 * hl);
      qa[qt][dc].q[1] = *reinterpret_cast<const uint4*>(Qb + dc * 32 + 8 * hl + 16);
    }
  }

  float mrun[2][8], lrun[2][8];
  CFrag acc[2][4];
#pragma unroll
  for (int qt = 0; qt < 2; ++qt) {
#pragma unroll
    for (int j = 0; j < 8; ++j) { mrun[qt][j] = -INFINITY; lrun[qt][j] = 0.0f; }
#pragma unroll
    for (int i = 0; i < 4; ++i)
#pragma unroll
      for (int j = 0; j < 8; ++j) acc[qt][i].f[j] = 0.0f;
  }

  const _Float16* Ktb = Kt + (size_t)head * DKH * S_LEN;
  const _Float16* Vb  = Vh + (size_t)head * S_LEN * DKH;

  stage_kv(Ks[0], Vs[0], Ktb, Vb, 0, t);          // prime stage 0

  for (int kb = 0; kb < S_LEN; kb += 32) {
    const int s = (kb >> 5) & 1;
    if (kb + 32 < S_LEN) {                        // issue next tile early
      stage_kv(Ks[s ^ 1], Vs[s ^ 1], Ktb, Vb, kb + 32, t);
#ifdef USE_ASYNC_LDS
      __builtin_amdgcn_s_wait_asynccnt(4);        // stage s (first 4) done
#endif
    } else {
#ifdef USE_ASYNC_LDS
      __builtin_amdgcn_s_wait_asynccnt(0);
#endif
    }
    __syncthreads();                              // tile s visible to block

    const _Float16* KsS = Ks[s];
    const _Float16* VsS = Vs[s];

    // Scores: for each key half (cols 0-15/16-31) and d-chunk, one B-frag
    // feeds both query tiles.
    CFrag sc[2][2];
#pragma unroll
    for (int qt = 0; qt < 2; ++qt)
#pragma unroll
      for (int kc = 0; kc < 2; ++kc)
#pragma unroll
        for (int j = 0; j < 8; ++j) sc[qt][kc].f[j] = 0.0f;

#pragma unroll
    for (int kc = 0; kc < 2; ++kc) {
#pragma unroll
      for (int dc = 0; dc < 2; ++dc) {
        AFrag b;  // lane L: col = key kc*16+L%16, K(=d) = dc*32 + 8*hl + (e&7) + 16*(e>>3)
        const _Float16* bp = &KsS[(dc * 32) * 40 + kc * 16 + ml];
        for (int e = 0; e < 16; ++e) b.h[e] = bp[(8 * hl + (e & 7) + 16 * (e >> 3)) * 40];
        sc[0][kc].v = wmma_f16(qa[0][dc].v, b.v, sc[0][kc].v);
        sc[1][kc].v = wmma_f16(qa[1][dc].v, b.v, sc[1][kc].v);
      }
    }

    // Online softmax per query tile; row M=j+8*hl spans a 16-lane half
#pragma unroll
    for (int qt = 0; qt < 2; ++qt) {
      _Float16* Pq = &Ps[w][qt][0];
#pragma unroll
      for (int j = 0; j < 8; ++j) {
        float red = fmaxf(sc[qt][0].f[j], sc[qt][1].f[j]);
#pragma unroll
        for (int o = 1; o < 16; o <<= 1) red = fmaxf(red, __shfl_xor(red, o, 32));
        float mnew  = fmaxf(mrun[qt][j], red);
        float scale = __expf(mrun[qt][j] - mnew);
        float p0 = __expf(sc[qt][0].f[j] - mnew);
        float p1 = __expf(sc[qt][1].f[j] - mnew);
        float rs = p0 + p1;
#pragma unroll
        for (int o = 1; o < 16; o <<= 1) rs += __shfl_xor(rs, o, 32);
        lrun[qt][j] = lrun[qt][j] * scale + rs;
        mrun[qt][j] = mnew;
        acc[qt][0].f[j] *= scale; acc[qt][1].f[j] *= scale;
        acc[qt][2].f[j] *= scale; acc[qt][3].f[j] *= scale;
        int prow = j + 8 * hl;      // restage P: C layout -> row-major LDS
        Pq[prow * 40 + ml]      = (_Float16)p0;
        Pq[prow * 40 + 16 + ml] = (_Float16)p1;
      }
    }

    // DS ops are in-order within a wave: read P back as A-fragments
    AFrag pa[2];
#pragma unroll
    for (int qt = 0; qt < 2; ++qt) {
      const _Float16* pp = &Ps[w][qt][ml * 40 + 8 * hl];
      pa[qt].q[0] = *reinterpret_cast<const uint4*>(pp);
      pa[qt].q[1] = *reinterpret_cast<const uint4*>(pp + 16);
    }

#pragma unroll
    for (int nt = 0; nt < 4; ++nt) {  // ctx += P(16x32) . V(32x64); V shared
      AFrag vb;
      const _Float16* vp = &VsS[nt * 16 + ml];
      for (int e = 0; e < 16; ++e) vb.h[e] = vp[(8 * hl + (e & 7) + 16 * (e >> 3)) * 72];
      acc[0][nt].v = wmma_f16(pa[0].v, vb.v, acc[0][nt].v);
      acc[1][nt].v = wmma_f16(pa[1].v, vb.v, acc[1][nt].v);
    }
    __syncthreads();   // block done with tile s before it is overwritten
  }

  // Normalize and write ctx (f16) in [S][768] (head-concat) layout
#pragma unroll
  for (int qt = 0; qt < 2; ++qt) {
#pragma unroll
    for (int nt = 0; nt < 4; ++nt) {
#pragma unroll
      for (int j = 0; j < 8; ++j) {
        int srow = m0 + (w * 2 + qt) * 16 + j + 8 * hl;
        float val = acc[qt][nt].f[j] / lrun[qt][j];
        Ctx[(size_t)srow * D_MODEL + head * DKH + nt * 16 + ml] = (_Float16)val;
      }
    }
  }
}

// ---------------------------------------------------------------------------
// Output projection: out_f32 = ctx_f16 @ Wo_f32 + bo.
// ---------------------------------------------------------------------------
__global__ __launch_bounds__(128) void proj_out_kernel(
    const _Float16* __restrict__ X, const float* __restrict__ W,
    const float* __restrict__ bias, float* __restrict__ Y) {
  __shared__ alignas(16) _Float16 Bs[32 * 72];

  const int m0 = blockIdx.x * 64;
  const int n0 = blockIdx.y * 64;
  const int t = threadIdx.x, lane = t & 31, w = t >> 5;
  const int ml = lane & 15, hl = lane >> 4;
  const int mrow = m0 + w * 16 + ml;

  CFrag c[4];
#pragma unroll
  for (int i = 0; i < 4; ++i)
#pragma unroll
    for (int j = 0; j < 8; ++j) c[i].f[j] = 0.0f;

  for (int k0 = 0; k0 < D_MODEL; k0 += 32) {
    __syncthreads();
#pragma unroll
    for (int i = 0; i < 2; ++i) {
      int u = t + 128 * i;
      int r = u >> 3, c8 = (u & 7) * 8;
      AFrag tmp;
      cvt8(tmp.h, W + (size_t)(k0 + r) * D_MODEL + n0 + c8);
      *reinterpret_cast<uint4*>(&Bs[r * 72 + c8]) = tmp.q[0];
    }
    __syncthreads();

    AFrag a;  // f16 source: direct b128 loads
    const _Float16* Xr = X + (size_t)mrow * D_MODEL + k0 + 8 * hl;
    a.q[0] = *reinterpret_cast<const uint4*>(Xr);
    a.q[1] = *reinterpret_cast<const uint4*>(Xr + 16);

#pragma unroll
    for (int nt = 0; nt < 4; ++nt) {
      AFrag b;
      const _Float16* bp = &Bs[nt * 16 + ml];
      for (int e = 0; e < 16; ++e) b.h[e] = bp[(8 * hl + (e & 7) + 16 * (e >> 3)) * 72];
      c[nt].v = wmma_f16(a.v, b.v, c[nt].v);
    }
  }

#pragma unroll
  for (int nt = 0; nt < 4; ++nt) {
    int n = n0 + nt * 16 + ml;
    float bv = bias[n];
#pragma unroll
    for (int j = 0; j < 8; ++j) {
      int srow = m0 + w * 16 + j + 8 * hl;
      Y[(size_t)srow * D_MODEL + n] = c[nt].f[j] + bv;
    }
  }
}

// ---------------------------------------------------------------------------
extern "C" void kernel_launch(void* const* d_in, const int* in_sizes, int n_in,
                              void* d_out, int out_size, void* d_ws,
                              size_t ws_size, hipStream_t stream) {
  (void)in_sizes; (void)n_in; (void)out_size; (void)ws_size;
  const float* q  = (const float*)d_in[0];
  const float* k  = (const float*)d_in[1];
  const float* v  = (const float*)d_in[2];
  const float* Wq = (const float*)d_in[3];
  const float* bq = (const float*)d_in[4];
  const float* Wk = (const float*)d_in[5];
  const float* bk = (const float*)d_in[6];
  const float* Wv = (const float*)d_in[7];
  const float* bv = (const float*)d_in[8];
  const float* Wo = (const float*)d_in[9];
  const float* bo = (const float*)d_in[10];
  float* out = (float*)d_out;

  const size_t headElems = (size_t)NH * S_LEN * DKH;  // 3,145,728 halves
  _Float16* Qh  = (_Float16*)d_ws;
  _Float16* Kt  = Qh + headElems;
  _Float16* Vh  = Kt + headElems;
  _Float16* Ctx = Vh + headElems;   // total ws use ~25.2 MB

  dim3 grid(S_LEN / 64, D_MODEL / 64);  // (64, 12)
  dim3 blk(128);
  // Q scaled by 1/sqrt(DK)=1/8 so attention needs no extra scale
  proj_f16_kernel<<<grid, blk, 0, stream>>>(q, Wq, bq, Qh, 0.125f, 0);
  proj_f16_kernel<<<grid, blk, 0, stream>>>(k, Wk, bk, Kt, 1.0f, 1);
  proj_f16_kernel<<<grid, blk, 0, stream>>>(v, Wv, bv, Vh, 1.0f, 0);
  attn_kernel<<<dim3(S_LEN / 128, NH), blk, 0, stream>>>(Qh, Kt, Vh, Ctx);
  proj_out_kernel<<<grid, blk, 0, stream>>>(Ctx, Wo, bo, out);
}
